// ProceduralMemory_68066641707191
// MI455X (gfx1250) — matrix operand order repaired and checked by reading, not compiled, not run on hardware
//
#include <hip/hip_runtime.h>

#define DIM    1024
#define NSK    50
#define KSTEP  32          // WMMA bf16 K depth
#define CH     5           // experts per round (10 rounds)

// ---- fallback path (fp32 weights staged in LDS) ----
#define NT_S   64
#define AHS_S  1032
#define WTS_S  40

typedef __attribute__((ext_vector_type(16))) __bf16 v16bf;
typedef __attribute__((ext_vector_type(8)))  float  v8f;

union FragCast { uint4 q[2]; v16bf v; };

__device__ __forceinline__ unsigned short f2bf(float f) {
  unsigned int u = __builtin_bit_cast(unsigned int, f);
  u += 0x7FFFu + ((u >> 16) & 1u);      // RNE to bf16
  return (unsigned short)(u >> 16);
}

// WGP-scope prefetch of the next 128B cache line (scope 0 fills all levels;
// SE/DEV-scope prefetches only warm GL2, useless for GL2-hot data).
__device__ __forceinline__ void prefetch_line(const void* p) {
  asm volatile("global_prefetch_b8 %0, off offset:128" :: "v"(p));
}

// ---------------- gating: gates[n,e] = softmax(x @ Wg + bg) ----------------
__global__ void gates_kernel(const float* __restrict__ x,
                             const float* __restrict__ Wg,
                             const float* __restrict__ bg,
                             float* __restrict__ gates) {
  __shared__ float xs[DIM];
  __shared__ float part[4][64];
  __shared__ float lsh[NSK];
  __shared__ float red[2];
  const int n = blockIdx.x;
  const int t = threadIdx.x;                   // 256 threads
  const float4* xrow = (const float4*)(x + (size_t)n * DIM);
  for (int i = t; i < DIM / 4; i += 256) {
    float4 v = xrow[i];
    xs[4*i+0] = v.x; xs[4*i+1] = v.y; xs[4*i+2] = v.z; xs[4*i+3] = v.w;
  }
  __syncthreads();
  const int p = t >> 6;
  const int e = t & 63;
  float acc = 0.f;
  if (e < NSK) {
    const int d0 = p * (DIM / 4);
    for (int d = 0; d < DIM / 4; ++d)
      acc += xs[d0 + d] * Wg[(size_t)(d0 + d) * NSK + e];
  }
  part[p][e] = acc;
  __syncthreads();
  if (t < NSK) lsh[t] = part[0][t] + part[1][t] + part[2][t] + part[3][t] + bg[t];
  __syncthreads();
  if (t == 0) {
    float m = lsh[0];
    for (int i = 1; i < NSK; ++i) m = fmaxf(m, lsh[i]);
    float s = 0.f;
    for (int i = 0; i < NSK; ++i) s += __expf(lsh[i] - m);
    red[0] = m; red[1] = 1.f / s;
  }
  __syncthreads();
  if (t < NSK) gates[(size_t)n * NSK + t] = __expf(lsh[t] - red[0]) * red[1];
}

// ---------------- pre-pass: x fp32 -> bf16 ----------------
__global__ void cvt_x_kernel(const float* __restrict__ x,
                             unsigned short* __restrict__ xbf, int total4) {
  const int i = blockIdx.x * 256 + threadIdx.x;      // float4 index
  if (i < total4) {
    float4 v = ((const float4*)x)[i];
    unsigned long long pk =
        (unsigned long long)f2bf(v.x)         | ((unsigned long long)f2bf(v.y) << 16)
      | ((unsigned long long)f2bf(v.z) << 32) | ((unsigned long long)f2bf(v.w) << 48);
    ((unsigned long long*)xbf)[i] = pk;
  }
}

// ---- pre-pass: W[e][k][c] fp32 -> WT[e][c][k] bf16 (transpose per expert) ----
__global__ void cvt_w_kernel(const float* __restrict__ W1,
                             const float* __restrict__ W2,
                             unsigned short* __restrict__ W1T,
                             unsigned short* __restrict__ W2T) {
  __shared__ unsigned short tile[64][65];
  const int z = blockIdx.z;                           // 0..99
  const float* src = (z < NSK) ? W1 + (size_t)z * DIM * DIM
                               : W2 + (size_t)(z - NSK) * DIM * DIM;
  unsigned short* dst = (z < NSK) ? W1T + (size_t)z * DIM * DIM
                                  : W2T + (size_t)(z - NSK) * DIM * DIM;
  const int r0 = blockIdx.x * 64, c0 = blockIdx.y * 64;
  const int t  = threadIdx.x;                         // 256 threads
  const int tc = t & 63, tr = t >> 6;
  for (int rp = 0; rp < 64; rp += 4)
    tile[rp + tr][tc] = f2bf(src[(size_t)(r0 + rp + tr) * DIM + c0 + tc]);
  __syncthreads();
  for (int rp = 0; rp < 64; rp += 4) {
    const int orow = rp + tr;
    dst[(size_t)(c0 + orow) * DIM + r0 + tc] = tile[tc][orow];
  }
}

// ---------------- out = gates @ b2_stack (folds bias-gate term) ----------------
__global__ void bias_init_kernel(const float* __restrict__ gates,
                                 const float* __restrict__ b2,
                                 float* __restrict__ out) {
  __shared__ float gl[NSK];
  const int n = blockIdx.x, t = threadIdx.x;          // 256 threads
  if (t < NSK) gl[t] = gates[(size_t)n * NSK + t];
  __syncthreads();
  float a0 = 0.f, a1 = 0.f, a2 = 0.f, a3 = 0.f;
  const int d0 = t * 4;
  for (int e = 0; e < NSK; ++e) {
    const float g = gl[e];
    float4 bv = *(const float4*)(b2 + (size_t)e * DIM + d0);
    a0 += g * bv.x; a1 += g * bv.y; a2 += g * bv.z; a3 += g * bv.w;
  }
  *(float4*)(out + (size_t)n * DIM + d0) = make_float4(a0, a1, a2, a3);
}

// ---- shared 64x64-per-wave inner loop: A,B straight from global bf16 ----
// All 8 fragments loaded up front each k-step (distinct registers) so the
// compiler can clause the 16 loads and overlap them with the 16 WMMAs via
// graduated s_wait_loadcnt.
__device__ __forceinline__ void gemm_k(const unsigned short* __restrict__ Abase,
                                       const unsigned short* __restrict__ Bbase,
                                       int row0, int col0, int L16, int half,
                                       int ksBeg, int ksEnd,
                                       v8f acc[4][4]) {
  for (int ks = ksBeg; ks < ksEnd; ++ks) {
    v16bf Bf[4], Af[4];
    #pragma unroll
    for (int ni = 0; ni < 4; ++ni) {
      const unsigned short* bp =
          Bbase + (size_t)(col0 + ni * 16 + L16) * DIM + ks * KSTEP + half * 16;
      prefetch_line(bp);
      FragCast fb;
      fb.q[0] = *(const uint4*)bp;            // K = khalf*16 .. +7
      fb.q[1] = *(const uint4*)(bp + 8);      // K = khalf*16+8 .. +15
      Bf[ni] = fb.v;
    }
    #pragma unroll
    for (int mi = 0; mi < 4; ++mi) {
      const unsigned short* ap =
          Abase + (size_t)(row0 + mi * 16 + L16) * DIM + ks * KSTEP + half * 16;
      prefetch_line(ap);
      FragCast fa;
      fa.q[0] = *(const uint4*)ap;            // K = khalf*16 .. +7
      fa.q[1] = *(const uint4*)(ap + 8);      // K = khalf*16+8 .. +15
      Af[mi] = fa.v;
    }
    #pragma unroll
    for (int mi = 0; mi < 4; ++mi)
      #pragma unroll
      for (int ni = 0; ni < 4; ++ni)
        acc[mi][ni] = __builtin_amdgcn_wmma_f32_16x16x32_bf16(
            false, Af[mi], false, Bf[ni], (short)0, acc[mi][ni], false, false);
  }
}

__device__ __forceinline__ void zero_acc(v8f acc[4][4]) {
  #pragma unroll
  for (int mi = 0; mi < 4; ++mi)
    #pragma unroll
    for (int ni = 0; ni < 4; ++ni)
      #pragma unroll
      for (int i = 0; i < 8; ++i) acc[mi][ni][i] = 0.f;
}

// ---- GEMM1: H[ce][n][h] = bf16( relu(x @ W1[e] + b1[e]) * gate[n,e] ) ----
// block 256 thr = 8 waves (4 row-groups x 2 col-groups), tile 256x128
__global__ void __launch_bounds__(256)
gemm1_kernel(const unsigned short* __restrict__ xbf,
             const unsigned short* __restrict__ W1T,
             const float* __restrict__ b1,
             const float* __restrict__ gates,
             unsigned short* __restrict__ Hc,
             int e0, int N) {
  const int t = threadIdx.x, lane = t & 31, w = t >> 5;
  const int rowg = w >> 1, colg = w & 1;
  const int ce = blockIdx.z, e = e0 + ce;
  const int row0 = blockIdx.x * 256 + rowg * 64;
  const int col0 = blockIdx.y * 128 + colg * 64;
  const int L16 = lane & 15, half = lane >> 4;
  const unsigned short* WB = W1T + (size_t)e * DIM * DIM;

  v8f acc[4][4];
  zero_acc(acc);
  gemm_k(xbf, WB, row0, col0, L16, half, 0, DIM / KSTEP, acc);

  unsigned short* Hb = Hc + (size_t)ce * N * DIM;
  #pragma unroll
  for (int mi = 0; mi < 4; ++mi) {
    float gv[8];
    #pragma unroll
    for (int i = 0; i < 8; ++i)
      gv[i] = gates[(size_t)(row0 + mi * 16 + i + half * 8) * NSK + e];
    #pragma unroll
    for (int ni = 0; ni < 4; ++ni) {
      const int col = col0 + ni * 16 + L16;
      const float bb = b1[(size_t)e * DIM + col];
      #pragma unroll
      for (int i = 0; i < 8; ++i) {
        const int row = row0 + mi * 16 + i + half * 8;
        float hv = fmaxf(acc[mi][ni][i] + bb, 0.f) * gv[i];
        Hb[(size_t)row * DIM + col] = f2bf(hv);
      }
    }
  }
}

// ---- GEMM2: out[n,d] += sum_ce H[ce] @ W2[e0+ce] ----
// block 256 thr = 8 waves: 4 spatial (2x2, 64x64 tiles) x 2 K-groups;
// K-groups reduced through LDS; single owner per out tile -> no atomics.
__global__ void __launch_bounds__(256)
gemm2_kernel(const unsigned short* __restrict__ Hc,
             const unsigned short* __restrict__ W2T,
             float* __restrict__ out,
             int e0, int N) {
  extern __shared__ char smem2[];
  float* red = (float*)smem2;                  // 4 waves * 4096 floats = 64 KB
  const int t = threadIdx.x, lane = t & 31, w = t >> 5;
  const int kg = w >> 2;                       // K-group 0/1
  const int wsid = w & 3;
  const int rowg = wsid >> 1, colg = wsid & 1;
  const int row0 = blockIdx.x * 128 + rowg * 64;
  const int col0 = blockIdx.y * 128 + colg * 64;
  const int L16 = lane & 15, half = lane >> 4;

  const int KS_TOT = CH * (DIM / KSTEP);       // 160 k-steps
  const int kLo = kg * (KS_TOT / 2);
  const int kHi = kLo + (KS_TOT / 2);

  v8f acc[4][4];
  zero_acc(acc);
  for (int ce = 0; ce < CH; ++ce) {
    const int b  = max(kLo - ce * (DIM / KSTEP), 0);
    const int en = min(kHi - ce * (DIM / KSTEP), DIM / KSTEP);
    if (b < en) {
      const unsigned short* A  = Hc + (size_t)ce * N * DIM;
      const unsigned short* WB = W2T + (size_t)(e0 + ce) * DIM * DIM;
      gemm_k(A, WB, row0, col0, L16, half, b, en, acc);
    }
  }

  float* myred = red + wsid * 4096;
  if (kg == 1) {
    #pragma unroll
    for (int mi = 0; mi < 4; ++mi)
      #pragma unroll
      for (int ni = 0; ni < 4; ++ni)
        #pragma unroll
        for (int i = 0; i < 8; ++i)
          myred[((mi * 4 + ni) * 8 + i) * 32 + lane] = acc[mi][ni][i];
  }
  __syncthreads();
  if (kg == 0) {
    #pragma unroll
    for (int mi = 0; mi < 4; ++mi)
      #pragma unroll
      for (int ni = 0; ni < 4; ++ni) {
        const int col = col0 + ni * 16 + L16;
        #pragma unroll
        for (int i = 0; i < 8; ++i) {
          const int row = row0 + mi * 16 + i + half * 8;
          const float v = acc[mi][ni][i] + myred[((mi * 4 + ni) * 8 + i) * 32 + lane];
          float* op = out + (size_t)row * DIM + col;
          *op = *op + v;
        }
      }
  }
}

// =======================================================================
// FALLBACK PATH (small ws): Round-1 fused kernel, fp32 weights via LDS.
// =======================================================================
__device__ __forceinline__ void gemm_tile_s(const float* __restrict__ Wglob,
                                            unsigned short* AH, unsigned short* WT,
                                            int t, int lane, int colBase,
                                            v8f acc[4][4]) {
  const int L16  = lane & 15;
  const int half = lane >> 4;
  for (int ks = 0; ks < DIM / KSTEP; ++ks) {
    #pragma unroll
    for (int cc = 0; cc < 2; ++cc) {
      const int c = t + cc * 512;
      const float* wp = Wglob + (size_t)(ks * KSTEP) * DIM + c;
      unsigned int wpack[KSTEP / 2];
      #pragma unroll
      for (int kk = 0; kk < KSTEP / 2; ++kk) {
        unsigned int lo = f2bf(wp[(size_t)(2 * kk)     * DIM]);
        unsigned int hi = f2bf(wp[(size_t)(2 * kk + 1) * DIM]);
        wpack[kk] = lo | (hi << 16);
      }
      uint4* dst = (uint4*)(WT + (size_t)c * WTS_S);
      dst[0] = make_uint4(wpack[0],  wpack[1],  wpack[2],  wpack[3]);
      dst[1] = make_uint4(wpack[4],  wpack[5],  wpack[6],  wpack[7]);
      dst[2] = make_uint4(wpack[8],  wpack[9],  wpack[10], wpack[11]);
      dst[3] = make_uint4(wpack[12], wpack[13], wpack[14], wpack[15]);
    }
    __syncthreads();
    v16bf A[4];
    #pragma unroll
    for (int mi = 0; mi < 4; ++mi) {
      FragCast fc;
      const unsigned short* ap =
          AH + (size_t)(mi * 16 + L16) * AHS_S + ks * KSTEP + half * 16;
      fc.q[0] = *(const uint4*)ap;
      fc.q[1] = *(const uint4*)(ap + 8);
      A[mi] = fc.v;
    }
    #pragma unroll
    for (int ni = 0; ni < 4; ++ni) {
      FragCast fb;
      const unsigned short* bp =
          WT + (size_t)(colBase + ni * 16 + L16) * WTS_S + half * 16;
      fb.q[0] = *(const uint4*)bp;
      fb.q[1] = *(const uint4*)(bp + 8);
      const v16bf B = fb.v;
      #pragma unroll
      for (int mi = 0; mi < 4; ++mi)
        acc[mi][ni] = __builtin_amdgcn_wmma_f32_16x16x32_bf16(
            false, A[mi], false, B, (short)0, acc[mi][ni], false, false);
    }
    __syncthreads();
  }
}

__global__ void __launch_bounds__(512)
moe_slow_kernel(const float* __restrict__ x,
                const float* __restrict__ W1, const float* __restrict__ b1,
                const float* __restrict__ W2, const float* __restrict__ b2,
                const float* __restrict__ gates,
                float* __restrict__ out) {
  extern __shared__ char smem[];
  unsigned short* AH = (unsigned short*)smem;
  unsigned short* WT = (unsigned short*)(smem + (size_t)NT_S * AHS_S * 2);
  float* B1L = (float*)(smem + (size_t)NT_S * AHS_S * 2 + (size_t)DIM * WTS_S * 2);
  float* B2L = B1L + DIM;
  float* GL  = B2L + DIM;

  const int n0 = blockIdx.x * NT_S;
  const int e  = blockIdx.y;
  const int t  = threadIdx.x;
  const int lane = t & 31;
  const int w  = t >> 5;
  const int colBase = w * 64;
  const int L16  = lane & 15;
  const int half = lane >> 4;

  const float* W1e = W1 + (size_t)e * DIM * DIM;
  const float* W2e = W2 + (size_t)e * DIM * DIM;

  for (int i = t; i < DIM; i += 512) {
    B1L[i] = b1[(size_t)e * DIM + i];
    B2L[i] = b2[(size_t)e * DIM + i];
  }
  if (t < NT_S) GL[t] = gates[(size_t)(n0 + t) * NSK + e];
  for (int f = t; f < NT_S * DIM / 4; f += 512) {
    const int row = f >> 8;
    const int c4  = f & 255;
    float4 v = *(const float4*)(x + (size_t)(n0 + row) * DIM + 4 * c4);
    unsigned long long pk =
        (unsigned long long)f2bf(v.x)         | ((unsigned long long)f2bf(v.y) << 16)
      | ((unsigned long long)f2bf(v.z) << 32) | ((unsigned long long)f2bf(v.w) << 48);
    *(unsigned long long*)(AH + (size_t)row * AHS_S + 4 * c4) = pk;
  }
  __syncthreads();

  v8f acc[4][4];
  zero_acc(acc);
  gemm_tile_s(W1e, AH, WT, t, lane, colBase, acc);

  #pragma unroll
  for (int mi = 0; mi < 4; ++mi)
    #pragma unroll
    for (int ni = 0; ni < 4; ++ni) {
      const int col = colBase + ni * 16 + L16;
      const float bb = B1L[col];
      #pragma unroll
      for (int i = 0; i < 8; ++i) {
        const int row = mi * 16 + i + half * 8;
        float hv = fmaxf(acc[mi][ni][i] + bb, 0.f) * GL[row];
        AH[(size_t)row * AHS_S + col] = f2bf(hv);
      }
    }
  __syncthreads();

  zero_acc(acc);
  gemm_tile_s(W2e, AH, WT, t, lane, colBase, acc);

  #pragma unroll
  for (int mi = 0; mi < 4; ++mi)
    #pragma unroll
    for (int ni = 0; ni < 4; ++ni) {
      const int col = colBase + ni * 16 + L16;
      const float bb = B2L[col];
      #pragma unroll
      for (int i = 0; i < 8; ++i) {
        const int row = mi * 16 + i + half * 8;
        const float v = acc[mi][ni][i] + GL[row] * bb;
        unsafeAtomicAdd(&out[(size_t)(n0 + row) * DIM + col], v);
      }
    }
}

extern "C" void kernel_launch(void* const* d_in, const int* in_sizes, int n_in,
                              void* d_out, int out_size, void* d_ws, size_t ws_size,
                              hipStream_t stream) {
  const float* x  = (const float*)d_in[0];
  const float* W1 = (const float*)d_in[1];
  const float* b1 = (const float*)d_in[2];
  const float* W2 = (const float*)d_in[3];
  const float* b2 = (const float*)d_in[4];
  const float* Wg = (const float*)d_in[5];
  const float* bg = (const float*)d_in[6];
  float* out = (float*)d_out;
  const int N = in_sizes[0] / DIM;              // 4096 tokens

  // fast-path ws layout: [xbf][W1T][W2T][H][gates]
  // (every prefetched region is followed by another mapped region)
  const size_t xbfB   = (size_t)N * DIM * 2;
  const size_t wB     = (size_t)NSK * DIM * DIM * 2;
  const size_t hB     = (size_t)CH * N * DIM * 2;
  const size_t gatesB = (size_t)N * NSK * sizeof(float);
  const size_t w1Off  = xbfB;
  const size_t w2Off  = w1Off + wB;
  const size_t hOff   = w2Off + wB;
  const size_t gOff   = hOff + hB;
  const size_t need   = gOff + gatesB;          // ~250 MB

  if (ws_size >= need) {
    unsigned short* xbf   = (unsigned short*)d_ws;
    unsigned short* W1T   = (unsigned short*)((char*)d_ws + w1Off);
    unsigned short* W2T   = (unsigned short*)((char*)d_ws + w2Off);
    unsigned short* Hc    = (unsigned short*)((char*)d_ws + hOff);
    float*          gates = (float*)((char*)d_ws + gOff);

    gates_kernel<<<N, 256, 0, stream>>>(x, Wg, bg, gates);
    cvt_x_kernel<<<(N * DIM / 4 + 255) / 256, 256, 0, stream>>>(x, xbf, N * DIM / 4);
    dim3 gw(16, 16, 2 * NSK);
    cvt_w_kernel<<<gw, 256, 0, stream>>>(W1, W2, W1T, W2T);
    bias_init_kernel<<<N, 256, 0, stream>>>(gates, b2, out);   // out = gates @ b2

    dim3 g1(N / 256, DIM / 128, CH);
    dim3 g2(N / 128, DIM / 128);
    for (int e0 = 0; e0 < NSK; e0 += CH) {
      gemm1_kernel<<<g1, 256, 0, stream>>>(xbf, W1T, b1, gates, Hc, e0, N);
      gemm2_kernel<<<g2, 256, 65536, stream>>>(Hc, W2T, out, e0, N);
    }
  } else {
    float* gates = (float*)d_ws;                // slow path: gates at ws start
    gates_kernel<<<N, 256, 0, stream>>>(x, Wg, bg, gates);
    hipMemsetAsync(d_out, 0, (size_t)N * DIM * sizeof(float), stream);
    const size_t shmem = (size_t)NT_S * AHS_S * 2 + (size_t)DIM * WTS_S * 2
                       + 2 * DIM * sizeof(float) + NT_S * sizeof(float);
    dim3 grid(N / NT_S, NSK);
    moe_slow_kernel<<<grid, 512, shmem, stream>>>(x, W1, b1, W2, b2, gates, out);
  }
}
